// Attention_32830730010708
// MI455X (gfx1250) — matrix operand order repaired and checked by reading, not compiled, not run on hardware
//
#include <hip/hip_runtime.h>
#include <math.h>
#include <stdint.h>

typedef __attribute__((ext_vector_type(16))) _Float16 v16h;
typedef __attribute__((ext_vector_type(8)))  _Float16 v8h;
typedef __attribute__((ext_vector_type(8)))  float    v8f;
typedef __attribute__((ext_vector_type(4)))  uint32_t u32x4;
typedef __attribute__((ext_vector_type(8)))  uint32_t u32x8;

#define WMMA_F16(A,B,C) __builtin_amdgcn_wmma_f32_16x16x32_f16(false,(A),false,(B),(short)0,(C),false,false)

constexpr int Bsz  = 2;
constexpr int S    = 1024;
constexpr int HID  = 2048;
constexpr int NH   = 32;
constexpr int NKV  = 8;
constexpr int HD   = 64;
constexpr int Mrows = Bsz * S;        // 2048
constexpr int NQ    = NH * HD;        // 2048
constexpr int NKVD  = NKV * HD;       // 512

// Build a v16h A/B fragment from two contiguous 16-byte chunks.
__device__ __forceinline__ v16h ld16(const _Float16* p0, const _Float16* p1) {
  v8h lo = *(const v8h*)p0;
  v8h hi = *(const v8h*)p1;
  return __builtin_shufflevector(lo, hi, 0,1,2,3,4,5,6,7,8,9,10,11,12,13,14,15);
}

// Issue 64 bytes (4 x b128) of async global->LDS copy for this lane (ASYNCcnt).
__device__ __forceinline__ void async_copy64(uint32_t lds_off, const _Float16* gsrc) {
  asm volatile(
      "global_load_async_to_lds_b128 %0, %1, off\n\t"
      "global_load_async_to_lds_b128 %0, %1, off offset:16\n\t"
      "global_load_async_to_lds_b128 %0, %1, off offset:32\n\t"
      "global_load_async_to_lds_b128 %0, %1, off offset:48"
      :: "v"(lds_off), "v"(gsrc) : "memory");
}

// Issue one TDM 2D tile load (TENSORcnt). g1 carries the static tile geometry;
// group0 is rebuilt per call with the LDS destination and 57-bit global address.
__device__ __forceinline__ void tdm_load_tile(uint32_t lds_addr, const void* gptr,
                                              u32x8 g1) {
  uint64_t ga = (uint64_t)(uintptr_t)gptr;
  u32x4 g0;
  g0[0] = 1u;                                                   // count=1, user D#
  g0[1] = lds_addr;                                             // lds_addr[31:0]
  g0[2] = (uint32_t)ga;                                         // global_addr[31:0]
  g0[3] = ((uint32_t)(ga >> 32) & 0x01FFFFFFu) | 0x80000000u;   // addr[56:32] | type=2
  asm volatile("tensor_load_to_lds %0, %1" :: "s"(g0), "s"(g1) : "memory");
}

// ---------------- prep: fp32 -> f16 convert / transpose -----------------

__global__ void cvt_f16_kernel(const float* __restrict__ x, _Float16* __restrict__ y, int n) {
  int i = blockIdx.x * blockDim.x + threadIdx.x;
  if (i < n) y[i] = (_Float16)x[i];
}

// w[K][N] fp32 -> wt[N][K] f16  (coalesced reads)
__global__ void transpose_f16_kernel(const float* __restrict__ w, _Float16* __restrict__ wt,
                                     int K, int N) {
  int idx = blockIdx.x * blockDim.x + threadIdx.x;
  if (idx < K * N) {
    int k = idx / N, n = idx - k * N;
    wt[(size_t)n * K + k] = (_Float16)w[idx];
  }
}

// ---------------- kernel 1: fused QKV projection + RoPE -----------------
// Block = 4 waves sharing one 64-wide N strip (one head for Q/K). Weight
// strip staged 64-K-deep into LDS via per-lane async global->LDS (ASYNCcnt),
// double buffered; each wave computes a 16x64 output tile (4 accumulators).
// N-strips: 0..31 Q heads, 32..39 K kv heads, 40..47 V kv heads.
__global__ void __launch_bounds__(128) qkv_rope_kernel(
    const _Float16* __restrict__ Xh,
    const _Float16* __restrict__ WqT, const _Float16* __restrict__ WkT,
    const _Float16* __restrict__ WvT,
    _Float16* __restrict__ Qh, _Float16* __restrict__ Kh, _Float16* __restrict__ VTh) {
  __shared__ __align__(128) _Float16 Bt[2][64][64];   // 16 KB double-buffered B strip

  int tid  = threadIdx.x;
  int lane = tid & 31;
  int l16  = lane & 15, hl = lane >> 4;
  int w    = tid >> 5;                  // wave in block
  int ns   = blockIdx.x % 48;           // 64-wide N strip (shared by block)
  int mg   = blockIdx.x / 48;           // 0..31
  int mt   = mg * 4 + w;                // row tile (16 rows), 0..127

  const _Float16* WT; int ncol0;
  if (ns < 32)      { WT = WqT; ncol0 = ns * 64; }
  else if (ns < 40) { WT = WkT; ncol0 = (ns - 32) * 64; }
  else              { WT = WvT; ncol0 = (ns - 40) * 64; }

  // cooperative B staging: thread -> (col, 32-half part)
  int ccol  = tid >> 1;
  int cpart = tid & 1;
  const _Float16* gB  = WT + (size_t)(ncol0 + ccol) * HID + cpart * 32;
  uint32_t ldsB = (uint32_t)(uintptr_t)&Bt[0][ccol][cpart * 32];

  const _Float16* arow = Xh + (size_t)(mt * 16 + l16) * HID + (hl << 4);

  v8f c0 = {}, c1 = {}, c2 = {}, c3 = {};
  constexpr int NSTAGE = HID / 64;      // 32 stages of K=64

  async_copy64(ldsB, gB);               // stage 0
  for (int st = 0; st < NSTAGE; ++st) {
    if (st + 1 < NSTAGE) {
      async_copy64(ldsB + ((st + 1) & 1) * 8192u, gB + (st + 1) * 64);
      asm volatile("s_wait_asynccnt 4" ::: "memory");
    } else {
      asm volatile("s_wait_asynccnt 0" ::: "memory");
    }
    __syncthreads();

    int cur = st & 1;
    int k0  = st * 64;
#pragma unroll
    for (int kk = 0; kk < 64; kk += 32) {
      v16h a  = ld16(arow + k0 + kk, arow + k0 + kk + 8);
      v16h b0 = *(const v16h*)&Bt[cur][ 0 + l16][kk + (hl << 4)];
      v16h b1 = *(const v16h*)&Bt[cur][16 + l16][kk + (hl << 4)];
      v16h b2 = *(const v16h*)&Bt[cur][32 + l16][kk + (hl << 4)];
      v16h b3 = *(const v16h*)&Bt[cur][48 + l16][kk + (hl << 4)];
      c0 = WMMA_F16(a, b0, c0);
      c1 = WMMA_F16(a, b1, c1);
      c2 = WMMA_F16(a, b2, c2);
      c3 = WMMA_F16(a, b3, c3);
    }
    __syncthreads();                    // protect buffer before next overwrite
  }

  if (ns < 40) {
    // Q or K: apply RoPE then store [b][head][s][d] f16.
    float fa = __powf(10000.f, -(float)l16        * (1.0f / 32.f));
    float fb = __powf(10000.f, -(float)(16 + l16) * (1.0f / 32.f));
#pragma unroll
    for (int r = 0; r < 8; ++r) {
      int m  = mt * 16 + r + 8 * hl;
      int bb = m >> 10, s = m & (S - 1);
      float sa, ca, sb, cb;
      __sincosf((float)s * fa, &sa, &ca);
      __sincosf((float)s * fb, &sb, &cb);
      float x0 = c0[r], x1 = c1[r], x2 = c2[r], x3 = c3[r];
      float y0 = x0 * ca - x2 * sa;   // d < 32 : x*cos - x[d+32]*sin
      float y2 = x2 * ca + x0 * sa;   // d >= 32: x*cos + x[d-32]*sin
      float y1 = x1 * cb - x3 * sb;
      float y3 = x3 * cb + x1 * sb;
      _Float16* dst;
      if (ns < 32) dst = Qh + ((size_t)(bb * NH  + ns)        * S + s) * HD;
      else         dst = Kh + ((size_t)(bb * NKV + (ns - 32)) * S + s) * HD;
      dst[ 0 + l16] = (_Float16)y0;
      dst[16 + l16] = (_Float16)y1;
      dst[32 + l16] = (_Float16)y2;
      dst[48 + l16] = (_Float16)y3;
    }
  } else {
    // V: store transposed [b][kv][d][s] f16 so PV B-fragments are contiguous.
    int kv = ns - 40;
#pragma unroll
    for (int r = 0; r < 8; ++r) {
      int m  = mt * 16 + r + 8 * hl;
      int bb = m >> 10, s = m & (S - 1);
      _Float16* dst = VTh + (size_t)(bb * NKV + kv) * HD * S + s;
      dst[(size_t)( 0 + l16) * S] = (_Float16)c0[r];
      dst[(size_t)(16 + l16) * S] = (_Float16)c1[r];
      dst[(size_t)(32 + l16) * S] = (_Float16)c2[r];
      dst[(size_t)(48 + l16) * S] = (_Float16)c3[r];
    }
  }
}

// ---------------- kernel 2: causal flash attention (one wave / 16 q-rows) ----
__global__ void __launch_bounds__(128) flash_attn_kernel(
    const _Float16* __restrict__ Qh, const _Float16* __restrict__ Kh,
    const _Float16* __restrict__ VTh, _Float16* __restrict__ Ah) {
  __shared__ _Float16 Pb[4][16][32];   // per-wave private P staging (C-layout -> A-layout)

  int gw   = (blockIdx.x * blockDim.x + threadIdx.x) >> 5;
  int lane = threadIdx.x & 31;
  int l16  = lane & 15, hl = lane >> 4;
  int wslot = (threadIdx.x >> 5) & 3;

  int qt = gw & 63;          // q tile (16 rows)
  int h  = (gw >> 6) & 31;   // query head
  int bb = gw >> 11;         // batch
  int kv = h >> 2;           // GQA: 4 q heads per kv head
  int qbase = qt * 16;

  const _Float16* Qp = Qh  + ((size_t)(bb * NH  + h)  * S + qbase) * HD;
  const _Float16* Kp = Kh  +  (size_t)(bb * NKV + kv) * S * HD;
  const _Float16* Vp = VTh +  (size_t)(bb * NKV + kv) * HD * S;

  int koff = hl << 4;
  const _Float16* qrow = Qp + l16 * HD;
  v16h qlo = ld16(qrow +  0 + koff, qrow +  8 + koff);   // d 0..31
  v16h qhi = ld16(qrow + 32 + koff, qrow + 40 + koff);   // d 32..63

  v8f acc0 = {}, acc1 = {}, acc2 = {}, acc3 = {};
  float rowmax[8], rowsum[8];
#pragma unroll
  for (int r = 0; r < 8; ++r) { rowmax[r] = -3.0e38f; rowsum[r] = 0.f; }

  const float sc = 0.125f;   // 1/sqrt(64)
  for (int k0 = 0; k0 < qbase + 16; k0 += 32) {
    // ---- scores: S = Q(16x64) * K^T(64x32), two 16-col subtiles ----
    const _Float16* kc0 = Kp + (size_t)(k0 +      l16) * HD + (hl << 4);
    const _Float16* kc1 = Kp + (size_t)(k0 + 16 + l16) * HD + (hl << 4);
    v16h kb00 = *(const v16h*)(kc0);        // cols k0..k0+15,  d 0..31
    v16h kb01 = *(const v16h*)(kc0 + 32);   //                  d 32..63
    v16h kb10 = *(const v16h*)(kc1);
    v16h kb11 = *(const v16h*)(kc1 + 32);
    v8f s0 = {}, s1 = {};
    s0 = WMMA_F16(qlo, kb00, s0);
    s0 = WMMA_F16(qhi, kb01, s0);
    s1 = WMMA_F16(qlo, kb10, s1);
    s1 = WMMA_F16(qhi, kb11, s1);

    // ---- online softmax (row = r + 8*hl, col = lane%16 within subtile) ----
#pragma unroll
    for (int r = 0; r < 8; ++r) {
      int q  = qbase + r + 8 * hl;
      float a  = s0[r] * sc;
      float b2 = s1[r] * sc;
      if (k0 +      l16 > q) a  = -3.0e38f;   // causal mask
      if (k0 + 16 + l16 > q) b2 = -3.0e38f;
      float cm = fmaxf(a, b2);
      cm = fmaxf(cm, __shfl_xor(cm, 1, 32));
      cm = fmaxf(cm, __shfl_xor(cm, 2, 32));
      cm = fmaxf(cm, __shfl_xor(cm, 4, 32));
      cm = fmaxf(cm, __shfl_xor(cm, 8, 32));
      float mnew  = fmaxf(rowmax[r], cm);
      float scale = __expf(rowmax[r] - mnew);
      rowmax[r] = mnew;
      float pa = __expf(a - mnew), pb = __expf(b2 - mnew);
      float ps = pa + pb;
      ps += __shfl_xor(ps, 1, 32);
      ps += __shfl_xor(ps, 2, 32);
      ps += __shfl_xor(ps, 4, 32);
      ps += __shfl_xor(ps, 8, 32);
      rowsum[r] = rowsum[r] * scale + ps;
      acc0[r] *= scale; acc1[r] *= scale; acc2[r] *= scale; acc3[r] *= scale;
      int mrow = r + 8 * hl;
      Pb[wslot][mrow][     l16] = (_Float16)pa;
      Pb[wslot][mrow][16 + l16] = (_Float16)pb;
    }
    asm volatile("s_wait_dscnt 0" ::: "memory");

    // ---- re-layout P into A-fragment, then O += P(16x32) * V(32x64) ----
    v16h pf = ld16(&Pb[wslot][l16][koff], &Pb[wslot][l16][8 + koff]);
    v16h vb0 = *(const v16h*)(Vp + (size_t)( 0 + l16) * S + k0 + (hl << 4));
    v16h vb1 = *(const v16h*)(Vp + (size_t)(16 + l16) * S + k0 + (hl << 4));
    v16h vb2 = *(const v16h*)(Vp + (size_t)(32 + l16) * S + k0 + (hl << 4));
    v16h vb3 = *(const v16h*)(Vp + (size_t)(48 + l16) * S + k0 + (hl << 4));
    acc0 = WMMA_F16(pf, vb0, acc0);
    acc1 = WMMA_F16(pf, vb1, acc1);
    acc2 = WMMA_F16(pf, vb2, acc2);
    acc3 = WMMA_F16(pf, vb3, acc3);
  }

  // normalize + store as [b][s][h*64+d] f16 (ready as O-proj A operand)
#pragma unroll
  for (int r = 0; r < 8; ++r) {
    float inv = 1.0f / rowsum[r];
    int q = qbase + r + 8 * hl;
    _Float16* dst = Ah + ((size_t)bb * S + q) * HID + h * HD;
    dst[ 0 + l16] = (_Float16)(acc0[r] * inv);
    dst[16 + l16] = (_Float16)(acc1[r] * inv);
    dst[32 + l16] = (_Float16)(acc2[r] * inv);
    dst[48 + l16] = (_Float16)(acc3[r] * inv);
  }
}

// ---------------- kernel 3: output projection (2048x2048x2048) -----------
// Weight strip staged with the Tensor Data Mover: each wave issues one 2D
// TDM descriptor per stage (64 contiguous K halves x 16 columns, row stride
// 2048), double buffered, synchronized with s_wait_tensorcnt + barrier.
__global__ void __launch_bounds__(128) oproj_kernel(
    const _Float16* __restrict__ Ah, const _Float16* __restrict__ WoT,
    float* __restrict__ out) {
  __shared__ __align__(128) _Float16 Bt[2][64][64];   // 16 KB

  int tid  = threadIdx.x;
  int lane = tid & 31;
  int l16  = lane & 15, hl = lane >> 4;
  int w    = tid >> 5;
  int ns   = blockIdx.x % 32;           // 64-wide N strip (shared by block)
  int mg   = blockIdx.x / 32;           // 0..31
  int mt   = mg * 4 + w;                // 0..127

  // Static TDM geometry (D# group 1): data_size=2B, tensor dims 64x16,
  // tile 64x16, dim0 stride = HID elements. One descriptor = 2KB tile.
  u32x8 g1;
  g1[0] = 0x00010000u;        // workgroup_mask=0 | data_size=1 (2 bytes)
  g1[1] = 64u << 16;          // tensor_dim0[15:0] = 64
  g1[2] = 16u << 16;          // tensor_dim0[31:16]=0 | tensor_dim1[15:0] = 16
  g1[3] = 64u << 16;          // tensor_dim1[31:16]=0 | tile_dim0 = 64
  g1[4] = 16u;                // tile_dim1 = 16 | tile_dim2 = 0
  g1[5] = (uint32_t)HID;      // tensor_dim0_stride[31:0] = 2048
  g1[6] = 0u;
  g1[7] = 0u;

  // this wave's quarter of the strip: columns [w*16, w*16+16)
  const _Float16* gW = WoT + (size_t)(ns * 64 + w * 16) * HID;
  uint32_t ldsW = (uint32_t)(uintptr_t)&Bt[0][w * 16][0];

  const _Float16* arow = Ah + (size_t)(mt * 16 + l16) * HID + (hl << 4);

  v8f c0 = {}, c1 = {}, c2 = {}, c3 = {};
  constexpr int NSTAGE = HID / 64;

  tdm_load_tile(ldsW, gW, g1);                        // stage 0
  for (int st = 0; st < NSTAGE; ++st) {
    if (st + 1 < NSTAGE) {
      tdm_load_tile(ldsW + ((st + 1) & 1) * 8192u, gW + (st + 1) * 64, g1);
      __builtin_amdgcn_s_wait_tensorcnt(1);           // stage st landed
    } else {
      __builtin_amdgcn_s_wait_tensorcnt(0);
    }
    __syncthreads();

    int cur = st & 1;
    int k0  = st * 64;
#pragma unroll
    for (int kk = 0; kk < 64; kk += 32) {
      v16h a  = ld16(arow + k0 + kk, arow + k0 + kk + 8);
      v16h b0 = *(const v16h*)&Bt[cur][ 0 + l16][kk + (hl << 4)];
      v16h b1 = *(const v16h*)&Bt[cur][16 + l16][kk + (hl << 4)];
      v16h b2 = *(const v16h*)&Bt[cur][32 + l16][kk + (hl << 4)];
      v16h b3 = *(const v16h*)&Bt[cur][48 + l16][kk + (hl << 4)];
      c0 = WMMA_F16(a, b0, c0);
      c1 = WMMA_F16(a, b1, c1);
      c2 = WMMA_F16(a, b2, c2);
      c3 = WMMA_F16(a, b3, c3);
    }
    __syncthreads();
  }

#pragma unroll
  for (int r = 0; r < 8; ++r) {
    int m = mt * 16 + r + 8 * hl;
    float* dst = out + (size_t)m * HID + ns * 64;
    dst[ 0 + l16] = c0[r];
    dst[16 + l16] = c1[r];
    dst[32 + l16] = c2[r];
    dst[48 + l16] = c3[r];
  }
}

// ---------------- host launch -----------------

extern "C" void kernel_launch(void* const* d_in, const int* in_sizes, int n_in,
                              void* d_out, int out_size, void* d_ws, size_t ws_size,
                              hipStream_t stream) {
  (void)in_sizes; (void)n_in; (void)out_size; (void)ws_size;
  const float* hs = (const float*)d_in[0];
  // d_in[1] = position_ids (arange, implicit), d_in[2] = additive causal mask
  // (synthesized in-kernel) — both encoded analytically per the reference.
  const float* Wq = (const float*)d_in[3];
  const float* Wk = (const float*)d_in[4];
  const float* Wv = (const float*)d_in[5];
  const float* Wo = (const float*)d_in[6];
  float* out = (float*)d_out;

  char* ws = (char*)d_ws;
  const size_t MB = 1ull << 20;
  _Float16* Xh  = (_Float16*)(ws +  0 * MB);   // [2048][2048]       8 MB
  _Float16* WqT = (_Float16*)(ws +  8 * MB);   // [2048][2048]       8 MB
  _Float16* WkT = (_Float16*)(ws + 16 * MB);   // [512][2048]        2 MB
  _Float16* WvT = (_Float16*)(ws + 18 * MB);   // [512][2048]        2 MB
  _Float16* WoT = (_Float16*)(ws + 20 * MB);   // [2048][2048]       8 MB
  _Float16* Qh  = (_Float16*)(ws + 28 * MB);   // [2][32][1024][64]  8 MB
  _Float16* Kh  = (_Float16*)(ws + 36 * MB);   // [2][8][1024][64]   2 MB
  _Float16* VTh = (_Float16*)(ws + 38 * MB);   // [2][8][64][1024]   2 MB
  _Float16* Ah  = (_Float16*)(ws + 40 * MB);   // [2048][2048]       8 MB

  int nx = Mrows * HID;
  cvt_f16_kernel<<<(nx + 255) / 256, 256, 0, stream>>>(hs, Xh, nx);
  transpose_f16_kernel<<<(HID * NQ   + 255) / 256, 256, 0, stream>>>(Wq, WqT, HID, NQ);
  transpose_f16_kernel<<<(HID * NKVD + 255) / 256, 256, 0, stream>>>(Wk, WkT, HID, NKVD);
  transpose_f16_kernel<<<(HID * NKVD + 255) / 256, 256, 0, stream>>>(Wv, WvT, HID, NKVD);
  transpose_f16_kernel<<<(HID * HID  + 255) / 256, 256, 0, stream>>>(Wo, WoT, HID, HID);

  // 48 strips * 32 m-groups, 4 waves/block (wave = one 16x64 tile)
  qkv_rope_kernel<<<1536, 128, 0, stream>>>(Xh, WqT, WkT, WvT, Qh, Kh, VTh);
  // 2*32*64 q-tiles = 4096 waves
  flash_attn_kernel<<<1024, 128, 0, stream>>>(Qh, Kh, VTh, Ah);
  // 32 strips * 32 m-groups
  oproj_kernel<<<1024, 128, 0, stream>>>(Ah, WoT, out);
}
